// W4Linear_81638738362814
// MI455X (gfx1250) — matrix operand order repaired and checked by reading, not compiled, not run on hardware
//
#include <hip/hip_runtime.h>
#include <hip/hip_bf16.h>
#include <stdint.h>

typedef __attribute__((ext_vector_type(16))) _Float16 v16h;
typedef __attribute__((ext_vector_type(8)))  _Float16 v8h;
typedef __attribute__((ext_vector_type(8)))  float    v8f;

#define KDIM 4096
#define NDIM 11008
#define MDIM 64
#define KCHUNK 512
#define NCHUNKS (KDIM / KCHUNK)          // 8
#define LDS_PITCH_H (KCHUNK + 8)         // 520 halves = 1040 B pitch (bank stagger)
#define NBLK 64                          // N columns per block (4 waves)

// ---------------------------------------------------------------------------
// Pre-kernel: x fp32 -> f16 (WMMA A operand) + per-row sums.
// rowsum folds the zero-point AND the +1024 nibble bias out of the GEMM loop.
// ---------------------------------------------------------------------------
__global__ __launch_bounds__(256) void w4_prep(const float* __restrict__ x,
                                               _Float16* __restrict__ xh,
                                               float* __restrict__ rowsum) {
  const int row = blockIdx.x;
  const int tid = threadIdx.x;
  const float* xr = x + (size_t)row * KDIM;
  _Float16* hr = xh + (size_t)row * KDIM;
  float s = 0.0f;
  for (int k = tid; k < KDIM; k += 256) {
    float v = xr[k];
    hr[k] = (_Float16)v;
    s += v;
  }
  __shared__ float red[256];
  red[tid] = s;
  __syncthreads();
  for (int off = 128; off > 0; off >>= 1) {
    if (tid < off) red[tid] += red[tid + off];
    __syncthreads();
  }
  if (tid == 0) rowsum[row] = red[0];
}

// ---------------------------------------------------------------------------
// GEMM: block = 4 waves = 64 N-cols x full M=64, K streamed through LDS.
// Analytic AWQ-unshuffle:
//   code(n, 32b + 8Q+2P+R) = nibble Q of qw[n>>2][(b>>1)*64 + (n&3)*16 + (b&1)*8 + 2P+R]
// Both nibbles a B-reg pair needs live in byte h of words W[2p],W[2p+1]:
//   t = v_perm(byte h of W[2p] -> byte0, byte h of W[2p+1] -> byte2)
//   Breg[p]   = 0x64006400 | ( t       & 0x000F000F)   // f16 = 1024+code (exact)
//   Breg[p+4] = 0x64006400 | ((t >> 4) & 0x000F000F)
// ---------------------------------------------------------------------------
__global__ __launch_bounds__(128) void w4_gemm(const int* __restrict__ qw,
                                               const _Float16* __restrict__ xh,
                                               const float* __restrict__ rowsum,
                                               const float* __restrict__ scale,
                                               const float* __restrict__ zp,
                                               const float* __restrict__ bias,
                                               float* __restrict__ out) {
  __shared__ _Float16 lds[MDIM * LDS_PITCH_H];

  const int tid  = threadIdx.x;
  const int wave = tid >> 5;
  const int lane = tid & 31;
  const int ln16 = lane & 15;
  const int h    = lane >> 4;            // half-wave id (K-half of fragments)

  const int n  = blockIdx.x * NBLK + wave * 16 + ln16;  // this lane's N column
  const int nq = n >> 2;                                 // packed weight row
  const int ii = n & 3;                                  // interleave slot

  // staging: 128 threads copy 64 rows x 512 halves; thread t -> row t&63, half t>>6
  const int crow  = tid & 63;
  const int chalf = tid >> 6;            // 0..1, 256 halves each

  v8f acc[4];
#pragma unroll
  for (int mt = 0; mt < 4; ++mt) { v8f z = {}; acc[mt] = z; }

  const uint32_t selp = 0x0C040C00u + (uint32_t)h * 0x00010001u; // byte h picks
  const int4* wrow = (const int4*)(qw + (size_t)nq * KDIM);

  for (int ch = 0; ch < NCHUNKS; ++ch) {
    // ---- stage x chunk (64 x 512 f16 = 64 KB) into LDS ----
    {
      const v8h* src = (const v8h*)(xh + (size_t)crow * KDIM + ch * KCHUNK + chalf * 256);
      v8h* dst = (v8h*)(lds + crow * LDS_PITCH_H + chalf * 256);
#pragma unroll
      for (int i = 0; i < 32; ++i) dst[i] = src[i];
    }
    __syncthreads();

    for (int b2 = 0; b2 < 8; ++b2) {      // K=64 per iteration (2 WMMA K-steps)
      const int t = ch * 8 + b2;
      const int4* wp = wrow + (t * 16 + ii * 4);   // (t*64 + ii*16) packed words
      union { int4 v[4]; uint32_t u[16]; } W;
      W.v[0] = wp[0]; W.v[1] = wp[1]; W.v[2] = wp[2]; W.v[3] = wp[3];
      if (t + 1 < KDIM / 64) __builtin_prefetch(wrow + ((t + 1) * 16 + ii * 4), 0, 1);

      union { v16h v; uint32_t u[8]; } Be, Bo;
#pragma unroll
      for (int p = 0; p < 4; ++p) {
        const uint32_t te = __builtin_amdgcn_perm(W.u[2 * p + 1], W.u[2 * p], selp);
        Be.u[p]     = 0x64006400u | (te & 0x000F000Fu);
        Be.u[p + 4] = 0x64006400u | ((te >> 4) & 0x000F000Fu);
        const uint32_t to = __builtin_amdgcn_perm(W.u[2 * p + 9], W.u[2 * p + 8], selp);
        Bo.u[p]     = 0x64006400u | (to & 0x000F000Fu);
        Bo.u[p + 4] = 0x64006400u | ((to >> 4) & 0x000F000Fu);
      }

      const int kb_e = b2 * 64 + h * 16;  // halves offset within chunk row
      const int kb_o = kb_e + 32;
#pragma unroll
      for (int mt = 0; mt < 4; ++mt) {
        const _Float16* ap = lds + (mt * 16 + ln16) * LDS_PITCH_H;
        union { v16h v; v8h p[2]; } Ae, Ao;
        Ae.p[0] = *(const v8h*)(ap + kb_e);
        Ae.p[1] = *(const v8h*)(ap + kb_e + 8);
        Ao.p[0] = *(const v8h*)(ap + kb_o);
        Ao.p[1] = *(const v8h*)(ap + kb_o + 8);
        acc[mt] = __builtin_amdgcn_wmma_f32_16x16x32_f16(
            false, Ae.v, false, Be.v, (short)0, acc[mt], false, false);
        acc[mt] = __builtin_amdgcn_wmma_f32_16x16x32_f16(
            false, Ao.v, false, Bo.v, (short)0, acc[mt], false, false);
      }
    }
    __syncthreads();
  }

  // ---- epilogue: out = scale*(acc - (zp+1024)*rowsum) + bias ----
  const float sc  = scale[n];
  const float zpp = zp[n] + 1024.0f;
  const float bi  = bias[n];
#pragma unroll
  for (int mt = 0; mt < 4; ++mt) {
#pragma unroll
    for (int e = 0; e < 8; ++e) {
      const int m = mt * 16 + h * 8 + e;  // C/D layout: lanes16-31 hold M=8..15
      out[(size_t)m * NDIM + n] = sc * (acc[mt][e] - zpp * rowsum[m]) + bi;
    }
  }
}

// ---------------------------------------------------------------------------
extern "C" void kernel_launch(void* const* d_in, const int* in_sizes, int n_in,
                              void* d_out, int out_size, void* d_ws, size_t ws_size,
                              hipStream_t stream) {
  const float* x     = (const float*)d_in[0];
  const int*   qw    = (const int*)d_in[1];
  const float* scale = (const float*)d_in[2];
  const float* zp    = (const float*)d_in[3];
  const float* bias  = (const float*)d_in[4];
  float* out = (float*)d_out;

  _Float16* xh  = (_Float16*)d_ws;
  float* rowsum = (float*)((char*)d_ws + (size_t)MDIM * KDIM * sizeof(_Float16));

  w4_prep<<<MDIM, 256, 0, stream>>>(x, xh, rowsum);
  w4_gemm<<<NDIM / NBLK, 128, 0, stream>>>(qw, xh, rowsum, scale, zp, bias, out);
}
